// Group_feat_network_33071248179251
// MI455X (gfx1250) — matrix operand order repaired and checked by reading, not compiled, not run on hardware
//
#include <hip/hip_runtime.h>
#include <hip/hip_bf16.h>

// Problem constants
#define NN 512
#define GG 60
#define KK 13

typedef __attribute__((ext_vector_type(16))) __bf16 v16bf;
typedef __attribute__((ext_vector_type(8)))  float  v8f;

union FragB { uint4 q[2]; v16bf v; };

__device__ __forceinline__ unsigned short f2bf(float f) {
  unsigned u = __float_as_uint(f);
  unsigned r = (u + 0x7FFFu + ((u >> 16) & 1u)) >> 16;  // round-to-nearest-even
  return (unsigned short)r;
}
__device__ __forceinline__ float bf2f(unsigned short h) {
  return __uint_as_float(((unsigned)h) << 16);
}

// feats [N,32,G] f32 -> A0 [N,G,32] bf16
__global__ __launch_bounds__(256) void cvt_feats(const float* __restrict__ F,
                                                 unsigned short* __restrict__ A0) {
  int idx = blockIdx.x * 256 + threadIdx.x;
  if (idx >= NN * GG * 32) return;
  int c = idx & 31;
  int g = (idx >> 5) % GG;
  int n = idx / (GG * 32);
  A0[idx] = f2bf(F[((size_t)n * 32 + c) * GG + g]);
}

// W [O,C,K] f32 -> Wc [O, k*C + c] bf16
__global__ __launch_bounds__(256) void cvt_w(const float* __restrict__ W,
                                             unsigned short* __restrict__ Wc,
                                             int cin, int total) {
  int idx = blockIdx.x * 256 + threadIdx.x;
  if (idx >= total) return;
  int kd = cin * KK;
  int o = idx / kd;
  int kk = idx - o * kd;
  int k = kk / cin;
  int c = kk - k * cin;
  Wc[idx] = f2bf(W[((size_t)o * cin + c) * KK + k]);
}

// Gather-GEMM: out[n,g,o] = sum_{k,c} X[n, nei[g,k], c] * Wc[o, k*CIN+c]  (+ fused epilogue)
// 256 threads = 8 waves arranged WAVES_M x WAVES_N; each wave owns a
// (WM_REG*16) x (WN_REG*16) register tile -> WM_REG*WN_REG v_wmma per K-step.
// Straight-line pipelined body (no branches): issue prefetch(s+1, clamped) ->
// 8x wmma on buf[s&1] -> wait + commit to buf[(s+1)&1] -> barrier.
template <int CIN, int COUT, int WAVES_M, int WAVES_N, int WM_REG, int WN_REG,
          bool RELU, bool ADD_RES, bool ADD_FEATS, bool OUT_F32>
__global__ __launch_bounds__(256) void comb_wmma(
    const unsigned short* __restrict__ X,    // [N, G, CIN] bf16
    const unsigned short* __restrict__ Wc,   // [COUT, 13*CIN] bf16
    const float* __restrict__ bias,          // [COUT]
    const int* __restrict__ nei,             // [G*13]
    const unsigned short* __restrict__ RES,  // [N, G, COUT] bf16 (L3 residual)
    const float* __restrict__ FRES,          // [N, 32, G] f32 (L4 residual = feats)
    unsigned short* __restrict__ Obf,        // [N, G, COUT] bf16
    float* __restrict__ Of32)                // [N, G, COUT] f32
{
  constexpr int WAVE_M = WM_REG * 16;
  constexpr int WAVE_N = WN_REG * 16;
  constexpr int MT     = WAVES_M * WAVE_M;  // block M tile
  constexpr int OT     = WAVES_N * WAVE_N;  // block O tile
  constexpr int CHUNKS = CIN / 32;          // 32-wide K chunks per neighbor
  constexpr int NSTEPS = KK * CHUNKS;
  constexpr int KD     = KK * CIN;
  constexpr int AS     = 40;                // LDS row stride (shorts), 16B-aligned rows
  static_assert(MT == 128, "staging path assumes 128-row block tile");
  __shared__ __align__(16) unsigned short As[2][MT * AS];
  __shared__ int s_nb[KK];                  // neighbor indices for this g

  const int g   = blockIdx.z;
  const int n0  = blockIdx.x * MT;
  const int o0  = blockIdx.y * OT;
  const int tid = threadIdx.x;
  const int lane = tid & 31;
  const int wave = tid >> 5;
  const int wm = wave / WAVES_N;
  const int wn = wave % WAVES_N;
  const int row = lane & 15;  // A row / B-D column within 16x16 tile
  const int kh  = lane >> 4;  // K-half select

  int ocol[WN_REG];
  const unsigned short* bptr[WN_REG];
#pragma unroll
  for (int j = 0; j < WN_REG; ++j) {
    ocol[j] = o0 + wn * WAVE_N + j * 16 + row;
    bptr[j] = Wc + (size_t)ocol[j] * KD + kh * 8;  // +s*32 per step, two 16B pieces at +0 and +16
  }

  const int srow = tid >> 2;        // staging rows 0..63 and 64..127
  const int scol = (tid & 3) * 8;   // 8 bf16 = 16B per thread per row
  const size_t xstride = (size_t)GG * CIN;
  // per-thread row bases for the gathered A tile
  const unsigned short* xrow0 = X + (size_t)(n0 + srow) * xstride + scol;
  const unsigned short* xrow1 = xrow0 + 64 * xstride;

  v8f acc[WM_REG][WN_REG];
#pragma unroll
  for (int i = 0; i < WM_REG; ++i)
#pragma unroll
    for (int j = 0; j < WN_REG; ++j)
#pragma unroll
      for (int v = 0; v < 8; ++v) acc[i][j][v] = 0.0f;

  if (tid < KK) s_nb[tid] = nei[g * KK + tid];

  // prologue: stage step 0 and publish it
  {
    const int nb = nei[g * KK];
    const uint4 d0 = *(const uint4*)(xrow0 + (size_t)nb * CIN);
    const uint4 d1 = *(const uint4*)(xrow1 + (size_t)nb * CIN);
    *(uint4*)(&As[0][srow * AS + scol]) = d0;
    *(uint4*)(&As[0][(srow + 64) * AS + scol]) = d1;
  }
  __syncthreads();

#pragma unroll 1
  for (int s = 0; s < NSTEPS; ++s) {
    // 1) unconditionally issue global prefetch for step s+1 (clamped on last iter)
    const int s1 = (s + 1 < NSTEPS) ? (s + 1) : s;
    const int k1 = s1 / CHUNKS;
    const int c1 = (s1 - k1 * CHUNKS) * 32;
    const int nb = s_nb[k1];
    const size_t poff = (size_t)nb * CIN + c1;
    const uint4 p0 = *(const uint4*)(xrow0 + poff);
    const uint4 p1 = *(const uint4*)(xrow1 + poff);

    // 2) compute on buf[s&1]
    const unsigned short* buf = As[s & 1];
    FragB fa[WM_REG], fb[WN_REG];
#pragma unroll
    for (int i = 0; i < WM_REG; ++i) {
      const unsigned short* ab = &buf[(wm * WAVE_M + i * 16 + row) * AS + kh * 8];
      fa[i].q[0] = *(const uint4*)(ab);
      fa[i].q[1] = *(const uint4*)(ab + 16);
    }
#pragma unroll
    for (int j = 0; j < WN_REG; ++j) {
      const unsigned short* bb = bptr[j] + s * 32;
      fb[j].q[0] = *(const uint4*)(bb);
      fb[j].q[1] = *(const uint4*)(bb + 16);
    }
#pragma unroll
    for (int i = 0; i < WM_REG; ++i)
#pragma unroll
      for (int j = 0; j < WN_REG; ++j)
        acc[i][j] = __builtin_amdgcn_wmma_f32_16x16x32_bf16(
            false, fa[i].v, false, fb[j].v, (short)0, acc[i][j], false, false);

    // 3) commit prefetched tile to the other buffer (wait lands here), then barrier
    unsigned short* nbuf = As[(s + 1) & 1];
    *(uint4*)(&nbuf[srow * AS + scol]) = p0;
    *(uint4*)(&nbuf[(srow + 64) * AS + scol]) = p1;
    __syncthreads();
  }

  // Epilogue: C/D layout -> lane col = row, M = v + kh*8. Fused bias/relu/residual.
#pragma unroll
  for (int j = 0; j < WN_REG; ++j) {
    const float bv = bias[ocol[j]];
#pragma unroll
    for (int i = 0; i < WM_REG; ++i) {
#pragma unroll
      for (int v = 0; v < 8; ++v) {
        const int m = n0 + wm * WAVE_M + i * 16 + kh * 8 + v;
        float val = acc[i][j][v] + bv;
        if (RELU) val = fmaxf(val, 0.0f);
        const size_t idx = (size_t)m * (GG * COUT) + (size_t)g * COUT + ocol[j];
        if (ADD_RES)   val += bf2f(RES[idx]);
        if (ADD_FEATS) val += FRES[(size_t)m * (32 * GG) + (size_t)ocol[j] * GG + g];
        if (OUT_F32) Of32[idx] = val;
        else         Obf[idx] = f2bf(val);
      }
    }
  }
}

// E [N,G,32] f32 (pre-normalization eqv). Per-n: channel norms per g, mean over g, inv norm.
__global__ __launch_bounds__(64) void finalize_kernel(const float* __restrict__ E,
                                                      float* __restrict__ out_inv,
                                                      float* __restrict__ out_eqv) {
  __shared__ float ssum[32];
  __shared__ float sden;
  const int n = blockIdx.x;
  const int t = threadIdx.x;
  if (t < 32) ssum[t] = 0.0f;
  __syncthreads();
  if (t < GG) {
    const float* e = E + ((size_t)n * GG + t) * 32;
    float v[32];
    float sq = 0.0f;
#pragma unroll
    for (int o = 0; o < 32; ++o) { v[o] = e[o]; sq += v[o] * v[o]; }
    const float rcp = 1.0f / fmaxf(sqrtf(sq), 1e-4f);
    float* oe = out_eqv + (size_t)n * 32 * GG + t;
#pragma unroll
    for (int o = 0; o < 32; ++o) oe[(size_t)o * GG] = v[o] * rcp;
#pragma unroll
    for (int o = 0; o < 32; ++o) atomicAdd(&ssum[o], v[o]);  // ds_add_f32
  }
  __syncthreads();
  if (t == 0) {
    float sq = 0.0f;
#pragma unroll
    for (int o = 0; o < 32; ++o) {
      float m = ssum[o] * (1.0f / GG);
      sq += m * m;
    }
    sden = fmaxf(sqrtf(sq), 1e-4f);
  }
  __syncthreads();
  if (t < 32) out_inv[(size_t)n * 32 + t] = (ssum[t] * (1.0f / GG)) / sden;
}

extern "C" void kernel_launch(void* const* d_in, const int* in_sizes, int n_in,
                              void* d_out, int out_size, void* d_ws, size_t ws_size,
                              hipStream_t stream) {
  const float* feats = (const float*)d_in[0];
  const int*   nei   = (const int*)d_in[1];
  const float* W_in  = (const float*)d_in[2];
  const float* b_in  = (const float*)d_in[3];
  const float* W_r1  = (const float*)d_in[4];
  const float* b_r1  = (const float*)d_in[5];
  const float* W_r2  = (const float*)d_in[6];
  const float* b_r2  = (const float*)d_in[7];
  const float* W_out = (const float*)d_in[8];
  const float* b_out = (const float*)d_in[9];

  char* ws = (char*)d_ws;
  size_t off = 0;
  auto alloc = [&](size_t bytes) -> char* {
    char* p = ws + off;
    off += (bytes + 255) & ~(size_t)255;
    return p;
  };

  unsigned short* A0 = (unsigned short*)alloc((size_t)NN * GG * 32 * 2);
  unsigned short* H  = (unsigned short*)alloc((size_t)NN * GG * 256 * 2);
  unsigned short* R  = (unsigned short*)alloc((size_t)NN * GG * 512 * 2);
  unsigned short* H2 = (unsigned short*)alloc((size_t)NN * GG * 256 * 2);
  float*          E  = (float*)alloc((size_t)NN * GG * 32 * 4);
  unsigned short* WcIn  = (unsigned short*)alloc((size_t)256 * (KK * 32) * 2);
  unsigned short* WcR1  = (unsigned short*)alloc((size_t)512 * (KK * 256) * 2);
  unsigned short* WcR2  = (unsigned short*)alloc((size_t)256 * (KK * 512) * 2);
  unsigned short* WcOut = (unsigned short*)alloc((size_t)32 * (KK * 256) * 2);
  (void)ws_size; (void)in_sizes; (void)n_in; (void)out_size;

  // Precision/layout conversion (cheap; weights stay hot in 192MB L2 afterwards)
  {
    int t = NN * GG * 32;
    cvt_feats<<<(t + 255) / 256, 256, 0, stream>>>(feats, A0);
  }
  { int t = 256 * KK * 32;  cvt_w<<<(t + 255) / 256, 256, 0, stream>>>(W_in,  WcIn,  32,  t); }
  { int t = 512 * KK * 256; cvt_w<<<(t + 255) / 256, 256, 0, stream>>>(W_r1,  WcR1,  256, t); }
  { int t = 256 * KK * 512; cvt_w<<<(t + 255) / 256, 256, 0, stream>>>(W_r2,  WcR2,  512, t); }
  { int t = 32 * KK * 256;  cvt_w<<<(t + 255) / 256, 256, 0, stream>>>(W_out, WcOut, 256, t); }

  dim3 blk(256);
  // Heavy layers: waves 2x4, register tile 4x2 -> block 128(M) x 128(O), 8 wmma/step/wave.
  // L1: 32 -> 256
  comb_wmma<32, 256, 2, 4, 4, 2, false, false, false, false>
      <<<dim3(NN / 128, 256 / 128, GG), blk, 0, stream>>>(A0, WcIn, b_in, nei,
                                                          nullptr, nullptr, H, nullptr);
  // L2: 256 -> 512, ReLU
  comb_wmma<256, 512, 2, 4, 4, 2, true, false, false, false>
      <<<dim3(NN / 128, 512 / 128, GG), blk, 0, stream>>>(H, WcR1, b_r1, nei,
                                                          nullptr, nullptr, R, nullptr);
  // L3: 512 -> 256, + h residual
  comb_wmma<512, 256, 2, 4, 4, 2, false, true, false, false>
      <<<dim3(NN / 128, 256 / 128, GG), blk, 0, stream>>>(R, WcR2, b_r2, nei,
                                                          H, nullptr, H2, nullptr);
  // L4: 256 -> 32, + original feats residual, f32 out (waves 8x1, tile 1x2 -> block 128x32)
  comb_wmma<256, 32, 8, 1, 1, 2, false, false, true, true>
      <<<dim3(NN / 128, 1, GG), blk, 0, stream>>>(H2, WcOut, b_out, nei,
                                                  nullptr, feats, nullptr, E);

  float* out_inv = (float*)d_out;
  float* out_eqv = out_inv + (size_t)NN * 32;
  finalize_kernel<<<NN, 64, 0, stream>>>(E, out_inv, out_eqv);
}
